// SemanticAlignmentModule_47115791237708
// MI455X (gfx1250) — matrix operand order repaired and checked by reading, not compiled, not run on hardware
//
#include <hip/hip_runtime.h>
#include <hip/hip_bf16.h>

// ---------------------------------------------------------------------------
// SemanticAlignmentModule for MI455X (gfx1250, wave32, WMMA)
//  - All GEMM operands pre-packed into the exact v_wmma_f32_16x16x32_bf16
//    register layout (fragment load = contiguous 32B = 2x b128).
//  - B tiles staged block-cooperatively into LDS with
//    global_load_async_to_lds_b128 (ASYNCcnt) in a double buffer, prefetching
//    k+1 while computing k; fragments then read via ds_load_b128 and shared
//    by all 8 waves of the block (8x less B traffic than per-wave loads).
// ---------------------------------------------------------------------------

typedef __attribute__((ext_vector_type(16))) __bf16 v16bf;
typedef __attribute__((ext_vector_type(8)))  float  v8f;

#define CDIM   512
#define KTOK   256
#define NTOK   4096             // H*W per batch
#define BATCH  8
#define TOTTOK (BATCH * NTOK)   // 32768
#define FFDIM  2048
#define TOPM   5
#define NEG_INF (-3.402823466e38f)

__device__ __forceinline__ float wave_sum(float v) {
#pragma unroll
  for (int d = 16; d >= 1; d >>= 1) v += __shfl_xor(v, d, 32);
  return v;
}

// ------------------------- packed fragment layouts -------------------------
// Packed A: matrix M x K tiled 16x32; block (mt,kt) = 32 lanes x 16 bf16.
// ISA 7.12.2 16-bit A layout: lane = (row&15) + 16*half, half = bit3 of k%32;
// within-lane slot = 2*jj + (k&1), jj = 4*(k%32>=16) + ((k%8)>>1).
__device__ __forceinline__ long packA_off(int row, int k, int nblkK) {
  const int kloc = k & 31;
  const int half = (kloc >> 3) & 1;
  const int jj   = ((kloc & 16) >> 2) + ((kloc & 7) >> 1);
  const int lane = (row & 15) + (half << 4);
  return (((long)(row >> 4) * nblkK + (k >> 5)) << 9) + (lane << 4) + 2 * jj + (kloc & 1);
}

// Packed B: matrix K x N tiled 32x16; block (nt,kt) = 32 lanes x 16 bf16.
// lane = (n&15) + 16*bit3(k%32); within-lane slot = 8*(k%32>=16) + k%8.
__device__ __forceinline__ long packB_off(int k, int n, int nblkK) {
  const int kloc = k & 31;
  const int lane = (n & 15) + (((kloc >> 3) & 1) << 4);
  return (((long)(n >> 4) * nblkK + (k >> 5)) << 9) + (lane << 4) + ((kloc >> 4) << 3) + (kloc & 7);
}

// --------------------------------- GEMM ------------------------------------
// Block = 8 waves = 128 rows x 64 cols. Per k-step the block async-copies the
// 4 B tiles (4KB) for k+1 into LDS (one b128 per thread) while computing k
// from the other LDS buffer; each wave issues 4 v_wmma_f32_16x16x32_bf16.
// EPI 0: f32 store (* alpha_mul [* exp(*alphap)])
// EPI 1: exact GELU + bias -> bf16 store in packed-A layout (feeds next GEMM)
// EPI 2: bias + residual add -> f32 store (final output)
template <int EPI>
__global__ __launch_bounds__(256) void gemm_wmma_bf16(
    const v16bf* __restrict__ Ap, int nblkK, long strideAblk,   // v16bf units
    const v16bf* __restrict__ Bp, long strideBblk,              // v16bf units
    float* __restrict__ Df, __bf16* __restrict__ Dh, int ldd, long strideD,
    const float* __restrict__ bias, const float* __restrict__ resid,
    float alpha_mul, const float* __restrict__ alphap) {
  __shared__ __align__(32) __bf16 smem[4096];   // 2 x 4KB B-strip double buffer

  const int tid   = threadIdx.x;
  const int lane  = tid & 31;
  const int wave  = tid >> 5;
  const int batch = blockIdx.z;
  const int mt  = blockIdx.y * 8 + wave;   // 16-row tile index
  const int nt0 = blockIdx.x * 4;          // first 16-col tile index
  Ap += (long)batch * strideAblk;
  Bp += (long)batch * strideBblk;

  const v16bf* __restrict__ arow = Ap + (((long)mt * nblkK) << 5) + lane;

  // Per-thread async-copy slice of the 4-tile B strip: thread t covers tile
  // (tid>>6), bytes (tid&63)*16.. +16 -> LDS byte tid*16 within the buffer.
  const char* Bbytes = (const char*)Bp;
  const long  gthr   = (((long)(nt0 + (tid >> 6)) * nblkK) << 10) + ((tid & 63) << 4);
  const unsigned lds0 = (unsigned)(uintptr_t)&smem[0] + (unsigned)(tid << 4);

#define ISSUE_B_PREFETCH(kt, buf)                                              \
  do {                                                                         \
    const unsigned long long g =                                               \
        (unsigned long long)(Bbytes + gthr + ((long)(kt) << 10));              \
    const unsigned l = lds0 + ((unsigned)(buf) << 12);                         \
    asm volatile("global_load_async_to_lds_b128 %0, %1, off"                   \
                 ::"v"(l), "v"(g) : "memory");                                 \
  } while (0)

  ISSUE_B_PREFETCH(0, 0);

  v8f acc0 = {}, acc1 = {}, acc2 = {}, acc3 = {};
  for (int kt = 0; kt < nblkK; ++kt) {
    asm volatile("s_wait_asynccnt 0x0" ::: "memory");  // own slice of buf[kt&1] done
    __syncthreads();                                   // all slices visible
    if (kt + 1 < nblkK) ISSUE_B_PREFETCH(kt + 1, (kt + 1) & 1);

    const v16bf a = arow[kt << 5];
    const v16bf* sb = (const v16bf*)smem + ((kt & 1) << 7) + lane;  // 128/buf
    acc0 = __builtin_amdgcn_wmma_f32_16x16x32_bf16(false, a, false, sb[0],
                                                   (short)0, acc0, false, false);
    acc1 = __builtin_amdgcn_wmma_f32_16x16x32_bf16(false, a, false, sb[32],
                                                   (short)0, acc1, false, false);
    acc2 = __builtin_amdgcn_wmma_f32_16x16x32_bf16(false, a, false, sb[64],
                                                   (short)0, acc2, false, false);
    acc3 = __builtin_amdgcn_wmma_f32_16x16x32_bf16(false, a, false, sb[96],
                                                   (short)0, acc3, false, false);
  }
#undef ISSUE_B_PREFETCH

  float alpha = alpha_mul;
  if (alphap) alpha *= __expf(*alphap);

  // C/D layout: VGPR r -> row = 16*mt + r + 8*(lane>=16), col = 16*nt + lane%16
  const int rbase = (mt << 4) + ((lane >> 4) << 3);
  const int cbase = (nt0 << 4) + (lane & 15);
  v8f accs[4] = {acc0, acc1, acc2, acc3};
#pragma unroll
  for (int t = 0; t < 4; ++t) {
#pragma unroll
    for (int r = 0; r < 8; ++r) {
      const int row = rbase + r;
      const int col = cbase + 16 * t;
      float v = accs[t][r] * alpha;
      if (bias) v += bias[col];
      if (EPI == 0) {
        Df[(long)batch * strideD + (long)row * ldd + col] = v;
      } else if (EPI == 1) {
        const float g = 0.5f * v * (1.0f + erff(v * 0.70710678118654752f));
        Dh[packA_off(row, col, ldd >> 5)] = (__bf16)g;   // packed-A for next GEMM
      } else {
        const long off = (long)row * ldd + col;
        Df[off] = v + resid[off];
      }
    }
  }
}

// ------------------------------- LayerNorm ---------------------------------
// One wave per token (C=512 -> 16 elems/lane); emits f32 row-major and
// packed-A bf16 (nblkK = 512/32 = 16).
__global__ __launch_bounds__(256) void ln_kernel(
    const float* __restrict__ in, float* __restrict__ outf,
    __bf16* __restrict__ outp, const float* __restrict__ g,
    const float* __restrict__ b) {
  const int lane = threadIdx.x & 31;
  const int wave = threadIdx.x >> 5;
  const long tok = (long)blockIdx.x * 8 + wave;
  const float* row = in + tok * CDIM;
  float x[16], s = 0.f, s2 = 0.f;
#pragma unroll
  for (int i = 0; i < 16; ++i) {
    x[i] = row[lane + 32 * i];
    s += x[i]; s2 += x[i] * x[i];
  }
  s = wave_sum(s); s2 = wave_sum(s2);
  const float mu   = s * (1.f / CDIM);
  const float var  = s2 * (1.f / CDIM) - mu * mu;
  const float rstd = rsqrtf(var + 1e-5f);
#pragma unroll
  for (int i = 0; i < 16; ++i) {
    const int c = lane + 32 * i;
    const float y = (x[i] - mu) * rstd * g[c] + b[c];
    outf[tok * CDIM + c] = y;
    outp[packA_off((int)tok, c, CDIM / 32)] = (__bf16)y;
  }
}

// text prep: one wave per text row; unit-norm row -> packed-B (k for q@k^T),
// raw row -> packed-A (text for text@Wv). Per-batch packed size = KTOK*CDIM.
__global__ __launch_bounds__(256) void text_prep_kernel(
    const float* __restrict__ text, __bf16* __restrict__ k_pack,
    __bf16* __restrict__ t_pack) {
  const int lane = threadIdx.x & 31;
  const int wave = threadIdx.x >> 5;
  const int row  = blockIdx.x * 8 + wave;       // [0, BATCH*KTOK)
  const int b    = row >> 8;                    // KTOK = 256
  const int kk   = row & (KTOK - 1);
  const float* p = text + (long)row * CDIM;
  const long pbase = (long)b * KTOK * CDIM;
  float x[16], s2 = 0.f;
#pragma unroll
  for (int i = 0; i < 16; ++i) {
    x[i] = p[lane + 32 * i];
    s2 += x[i] * x[i];
  }
  s2 = wave_sum(s2);
  const float rn = rsqrtf(s2);
#pragma unroll
  for (int i = 0; i < 16; ++i) {
    const int c = lane + 32 * i;
    k_pack[pbase + packB_off(c, kk, CDIM / 32)] = (__bf16)(x[i] * rn);
    t_pack[pbase + packA_off(kk, c, CDIM / 32)] = (__bf16)x[i];
  }
}

// q prep: one wave per token; unit-norm q row (f32) -> packed-A bf16.
__global__ __launch_bounds__(256) void q_prep_kernel(
    const float* __restrict__ q, __bf16* __restrict__ q_pack) {
  const int lane = threadIdx.x & 31;
  const int wave = threadIdx.x >> 5;
  const int row  = blockIdx.x * 8 + wave;
  const float* p = q + (long)row * CDIM;
  float x[16], s2 = 0.f;
#pragma unroll
  for (int i = 0; i < 16; ++i) {
    x[i] = p[lane + 32 * i];
    s2 += x[i] * x[i];
  }
  s2 = wave_sum(s2);
  const float rn = rsqrtf(s2);
#pragma unroll
  for (int i = 0; i < 16; ++i) {
    const int c = lane + 32 * i;
    q_pack[packA_off(row, c, CDIM / 32)] = (__bf16)(x[i] * rn);
  }
}

// weights: f32 row-major (k,n) -> packed-B bf16
__global__ __launch_bounds__(256) void pack_b_f32_kernel(
    const float* __restrict__ W, __bf16* __restrict__ dst, int Kdim, int N) {
  const long idx = (long)blockIdx.x * blockDim.x + threadIdx.x;
  if (idx >= (long)Kdim * N) return;
  const int k = (int)(idx / N);
  const int n = (int)(idx - (long)k * N);
  dst[packB_off(k, n, Kdim >> 5)] = (__bf16)W[idx];
}

// ---------------- top-5 + softmax + gather + gate + residual ----------------
// One wave per token. sim row = 256 floats (8/lane); iterative wave-argmax x5.
__global__ __launch_bounds__(256) void topk_attend_kernel(
    const float* __restrict__ sim, const float* __restrict__ vmat,
    const float* __restrict__ x, const float* __restrict__ Wg,
    const float* __restrict__ bg, float* __restrict__ y) {
  const int lane = threadIdx.x & 31;
  const int wave = threadIdx.x >> 5;
  const long tok = (long)blockIdx.x * 8 + wave;
  const int b = (int)(tok / NTOK);
  const float* srow = sim + tok * KTOK;

  float vals[8];
#pragma unroll
  for (int i = 0; i < 8; ++i) vals[i] = srow[i * 32 + lane];

  float tv[TOPM]; int ti[TOPM];
#pragma unroll
  for (int m = 0; m < TOPM; ++m) {
    float best = NEG_INF; int bi = 0x7fffffff;
#pragma unroll
    for (int i = 0; i < 8; ++i)
      if (vals[i] > best) { best = vals[i]; bi = i * 32 + lane; }
#pragma unroll
    for (int d = 16; d >= 1; d >>= 1) {
      const float ov = __shfl_xor(best, d, 32);
      const int   oi = __shfl_xor(bi, d, 32);
      if (ov > best || (ov == best && oi < bi)) { best = ov; bi = oi; }
    }
    tv[m] = best; ti[m] = bi;
    const int sl = bi >> 5, ln = bi & 31;
#pragma unroll
    for (int i = 0; i < 8; ++i)
      if (i == sl && lane == ln) vals[i] = NEG_INF;
  }

  float e[TOPM], den = 0.f;
#pragma unroll
  for (int m = 0; m < TOPM; ++m) { e[m] = __expf(tv[m] - tv[0]); den += e[m]; }
  const float rden = 1.f / den;

  const float* vb   = vmat + (long)b * KTOK * CDIM;
  const float* xrow = x + tok * CDIM;

  float gd = 0.f, xl[16];
#pragma unroll
  for (int j = 0; j < 16; ++j) {
    const int c = lane + 32 * j;
    xl[j] = xrow[c];
    gd += xl[j] * Wg[c];
  }
  gd = wave_sum(gd) + bg[0];
  const float gate = 1.f / (1.f + __expf(-gd));

#pragma unroll
  for (int j = 0; j < 16; ++j) {
    const int c = lane + 32 * j;
    float a = 0.f;
#pragma unroll
    for (int m = 0; m < TOPM; ++m)
      a += (e[m] * rden) * vb[(long)ti[m] * CDIM + c];
    y[tok * CDIM + c] = xl[j] + gate * a;
  }
}

// ------------------------------- launcher -----------------------------------
extern "C" void kernel_launch(void* const* d_in, const int* in_sizes, int n_in,
                              void* d_out, int out_size, void* d_ws, size_t ws_size,
                              hipStream_t stream) {
  const float* visual = (const float*)d_in[0];
  const float* text   = (const float*)d_in[1];
  const float* g1     = (const float*)d_in[2];
  const float* b1     = (const float*)d_in[3];
  const float* g2     = (const float*)d_in[4];
  const float* b2     = (const float*)d_in[5];
  const float* Wq     = (const float*)d_in[6];
  const float* Wv     = (const float*)d_in[7];
  const float* logit_scale = (const float*)d_in[8];
  const float* Wg     = (const float*)d_in[9];
  const float* bg     = (const float*)d_in[10];
  const float* W1     = (const float*)d_in[11];
  const float* bff1   = (const float*)d_in[12];
  const float* W2     = (const float*)d_in[13];
  const float* bff2   = (const float*)d_in[14];
  float* out = (float*)d_out;
  (void)ws_size; (void)n_in; (void)in_sizes; (void)out_size;

  // ---- workspace carve-up (256B aligned); x/q buffers recycled post-attn ----
  char* ws = (char*)d_ws;
  size_t off = 0;
  auto carve = [&](size_t bytes) -> char* {
    char* p = ws + off;
    off += (bytes + 255) & ~(size_t)255;
    return p;
  };
  float*  bufA   = (float*)carve((size_t)TOTTOK * CDIM * 4);   // x_f32 -> yln_f32
  float*  bufB   = (float*)carve((size_t)TOTTOK * CDIM * 4);   // q_f32 -> y_f32
  __bf16* bufC   = (__bf16*)carve((size_t)TOTTOK * CDIM * 2);  // xP -> ylnP (packed A)
  __bf16* q_p    = (__bf16*)carve((size_t)TOTTOK * CDIM * 2);  // packed A
  __bf16* k_p    = (__bf16*)carve((size_t)BATCH * KTOK * CDIM * 2);  // packed B /batch
  __bf16* t_p    = (__bf16*)carve((size_t)BATCH * KTOK * CDIM * 2);  // packed A /batch
  float*  v_f    = (float*)carve((size_t)BATCH * KTOK * CDIM * 4);
  float*  sim_f  = (float*)carve((size_t)TOTTOK * KTOK * 4);
  __bf16* h_p    = (__bf16*)carve((size_t)TOTTOK * FFDIM * 2); // packed A
  __bf16* Wq_p   = (__bf16*)carve((size_t)CDIM * CDIM * 2);    // packed B
  __bf16* Wv_p   = (__bf16*)carve((size_t)CDIM * CDIM * 2);
  __bf16* W1_p   = (__bf16*)carve((size_t)CDIM * FFDIM * 2);
  __bf16* W2_p   = (__bf16*)carve((size_t)FFDIM * CDIM * 2);

  const float inv_sqrt_c = 0.04419417382415922f;  // 1/sqrt(512)
  const long  PK = (long)KTOK * CDIM;             // per-batch packed elems

  // 0) weight packing (f32 row-major -> packed-B bf16)
  pack_b_f32_kernel<<<(CDIM * CDIM + 255) / 256, 256, 0, stream>>>(Wq, Wq_p, CDIM, CDIM);
  pack_b_f32_kernel<<<(CDIM * CDIM + 255) / 256, 256, 0, stream>>>(Wv, Wv_p, CDIM, CDIM);
  pack_b_f32_kernel<<<(CDIM * FFDIM + 255) / 256, 256, 0, stream>>>(W1, W1_p, CDIM, FFDIM);
  pack_b_f32_kernel<<<(FFDIM * CDIM + 255) / 256, 256, 0, stream>>>(W2, W2_p, FFDIM, CDIM);

  // 1) LN1(visual) -> x_f32 (bufA) + packed-A x (bufC)
  ln_kernel<<<TOTTOK / 8, 256, 0, stream>>>(visual, bufA, bufC, g1, b1);

  // 2) text prep -> packed-B k, packed-A text
  text_prep_kernel<<<(BATCH * KTOK) / 8, 256, 0, stream>>>(text, k_p, t_p);

  // 3) q = x @ Wq (32768x512, K=512) -> bufB f32
  gemm_wmma_bf16<0><<<dim3(CDIM / 64, TOTTOK / 128, 1), 256, 0, stream>>>(
      (const v16bf*)bufC, CDIM / 32, 0, (const v16bf*)Wq_p, 0,
      bufB, nullptr, CDIM, 0, nullptr, nullptr, 1.0f, nullptr);

  // 4) q row-normalize -> packed-A q_p
  q_prep_kernel<<<TOTTOK / 8, 256, 0, stream>>>(bufB, q_p);

  // 5) v = text @ Wv (batched 256x512, K=512) -> v_f f32 row-major
  gemm_wmma_bf16<0><<<dim3(CDIM / 64, KTOK / 128, BATCH), 256, 0, stream>>>(
      (const v16bf*)t_p, CDIM / 32, PK / 16, (const v16bf*)Wv_p, 0,
      v_f, nullptr, CDIM, PK, nullptr, nullptr, 1.0f, nullptr);

  // 6) sim = (q @ k^T) * exp(logit_scale)/sqrt(C)  (batched 4096x256, K=512)
  gemm_wmma_bf16<0><<<dim3(KTOK / 64, NTOK / 128, BATCH), 256, 0, stream>>>(
      (const v16bf*)q_p, CDIM / 32, (long)NTOK * CDIM / 16,
      (const v16bf*)k_p, PK / 16,
      sim_f, nullptr, KTOK, (long)NTOK * KTOK,
      nullptr, nullptr, inv_sqrt_c, logit_scale);

  // 7) top-5 / softmax / gather(v) / gate / residual -> y (bufB reuse)
  topk_attend_kernel<<<TOTTOK / 8, 256, 0, stream>>>(sim_f, v_f, bufA, Wg, bg, bufB);

  // 8) LN2(y) -> yln_f32 (bufA reuse) + packed-A yln (bufC reuse)
  ln_kernel<<<TOTTOK / 8, 256, 0, stream>>>(bufB, bufA, bufC, g2, b2);

  // 9) h = gelu(yln @ W1 + bff1) -> h_p (packed-A bf16, 32768x2048, K=512)
  gemm_wmma_bf16<1><<<dim3(FFDIM / 64, TOTTOK / 128, 1), 256, 0, stream>>>(
      (const v16bf*)bufC, CDIM / 32, 0, (const v16bf*)W1_p, 0,
      nullptr, h_p, FFDIM, 0, bff1, nullptr, 1.0f, nullptr);

  // 10) out = yln + h @ W2 + bff2 (32768x512, K=2048)
  gemm_wmma_bf16<2><<<dim3(CDIM / 64, TOTTOK / 128, 1), 256, 0, stream>>>(
      (const v16bf*)h_p, FFDIM / 32, 0, (const v16bf*)W2_p, 0,
      out, nullptr, CDIM, 0, bff2, bufA, 1.0f, nullptr);
}
